// ResettingGRU_54872502174038
// MI455X (gfx1250) — matrix operand order repaired and checked by reading, not compiled, not run on hardware
//
#include <hip/hip_runtime.h>
#include <math.h>

// ---------------- types ----------------
typedef __bf16 bf16;
typedef __attribute__((ext_vector_type(16))) __bf16 v16bf;
typedef __attribute__((ext_vector_type(8)))  __bf16 v8bf;
typedef __attribute__((ext_vector_type(8)))  float  v8f;
typedef __attribute__((ext_vector_type(4)))  float  v4f;

// ---------------- problem dims ----------------
#define BB   64
#define TT   512
#define DD   1024
#define HH   1024
#define G3H  3072

// ---------------- workspace layout (bytes) ----------------
#define WS_WIH  0
#define WS_WHH  6291456
#define WS_HBUF 12582912
#define WS_BAR  13107200
#define WS_GI   13107456

#define NWG_REC 64

// gfx1250 hardware tanh (TRANS op)
__device__ __forceinline__ float fast_tanh(float x) {
    float r;
    asm("v_tanh_f32 %0, %1" : "=v"(r) : "v"(x));
    return r;
}
__device__ __forceinline__ float fast_sig(float x) {
    return 0.5f * fast_tanh(0.5f * x) + 0.5f;
}

// async copy 16B global -> LDS (per lane), tracked by ASYNCcnt
__device__ __forceinline__ void async_b128(unsigned lds_off, const void* gaddr) {
    asm volatile("global_load_async_to_lds_b128 %0, %1, off"
                 :: "v"(lds_off), "v"(gaddr) : "memory");
}
__device__ __forceinline__ void wait_async0() {
    asm volatile("s_wait_asynccnt 0x0" ::: "memory");
}

// For an 8-aligned k-group inside a 32-wide K block, the 16-bit A-fragment
// order used here (contiguous, k = 16g + j) keeps lane-group g constant and j consecutive:
//   kk = k & 31:  g = kk>>4,  j0 = kk&8
// so 8 converted bf16 go out as one 16-byte ds_store.

// ---------------- init: fp32->bf16 weights, h0 copy, barrier reset ----------
__global__ void k_init(const float* __restrict__ Wih, const float* __restrict__ Whh,
                       const float* __restrict__ h0,
                       bf16* __restrict__ wih_b, bf16* __restrict__ whh_b,
                       float* __restrict__ hbuf0, unsigned* __restrict__ bar) {
    size_t i = (size_t)blockIdx.x * blockDim.x + threadIdx.x;
    size_t stride = (size_t)gridDim.x * blockDim.x;
    size_t n4 = (size_t)G3H * DD / 4;
    for (size_t k = i; k < n4; k += stride) {
        v4f a = ((const v4f*)Wih)[k];
        v4f b = ((const v4f*)Whh)[k];
        v8bf pa, pb;
#pragma unroll
        for (int u = 0; u < 4; ++u) { pa[u] = (bf16)a[u]; pb[u] = (bf16)b[u]; }
        // store 4 bf16 (low half of v8bf) = 8 bytes
        *(uint64_t*)&wih_b[k * 4] = ((uint64_t*)&pa)[0];
        *(uint64_t*)&whh_b[k * 4] = ((uint64_t*)&pb)[0];
    }
    for (size_t k = i; k < (size_t)BB * HH / 4; k += stride)
        ((v4f*)hbuf0)[k] = ((const v4f*)h0)[k];
    if (i == 0) *bar = 0u;
}

// ---------------- GEMM1: gi[t][b][n] = x[b,t,:] . Wih[n,:] + bih[n] ---------
// grid (48, 512), 128 threads (4 waves); wave w: M rows [m0+16w,+16) x 64 cols
__global__ __launch_bounds__(128) void k_gemm_gi(
    const float* __restrict__ x, const bf16* __restrict__ wih,
    const float* __restrict__ bih, float* __restrict__ gi) {
    __shared__ bf16 lds_a[4 * 512];   // 4 m-tiles, fragment-ready

    const int tid  = threadIdx.x;
    const int lane = tid & 31;
    const int wv   = tid >> 5;
    const int n0   = blockIdx.x * 64;
    const int m0   = blockIdx.y * 64;

    v8f acc[4];
#pragma unroll
    for (int i = 0; i < 4; ++i) acc[i] = (v8f)0.0f;

    for (int k0 = 0; k0 < DD; k0 += 32) {
        __syncthreads();
        // stage A 64x32: 8 elements/chunk, 2 chunks/thread
#pragma unroll
        for (int e = 0; e < 2; ++e) {
            int q = e * 128 + tid;          // 0..255
            int m = q >> 2;                 // 0..63
            int k = (q & 3) * 8;            // 0,8,16,24
            const float* xp = &x[(size_t)(m0 + m) * DD + k0 + k];
            v4f a0 = *(const v4f*)xp;
            v4f a1 = *(const v4f*)(xp + 4);
            int g  = k >> 4;
            int j0 = k & 8;
            v8bf pk;
#pragma unroll
            for (int u = 0; u < 4; ++u) { pk[u] = (bf16)a0[u]; pk[u + 4] = (bf16)a1[u]; }
            *(v8bf*)&lds_a[(m >> 4) * 512 + ((m & 15) + 16 * g) * 16 + j0] = pk;
        }
        __syncthreads();

        v16bf afrag = *(const v16bf*)&lds_a[wv * 512 + lane * 16];
#pragma unroll
        for (int nt = 0; nt < 4; ++nt) {
            const bf16* bp = &wih[(size_t)(n0 + nt * 16 + (lane & 15)) * DD
                                  + k0 + 16 * (lane >> 4)];
            if (k0 + 32 < DD) __builtin_prefetch(bp + 32, 0, 1);
            v16bf bfrag = *(const v16bf*)bp;
            acc[nt] = __builtin_amdgcn_wmma_f32_16x16x32_bf16(
                false, afrag, false, bfrag, (short)0, acc[nt], false, false);
        }
    }

#pragma unroll
    for (int nt = 0; nt < 4; ++nt) {
        int n = n0 + nt * 16 + (lane & 15);
        float bias = bih[n];
#pragma unroll
        for (int i = 0; i < 8; ++i) {
            int m = m0 + wv * 16 + i + 8 * (lane >> 4);
            int t = m & (TT - 1);
            int b = m >> 9;
            gi[(size_t)t * (BB * G3H) + (size_t)b * G3H + n] = acc[nt][i] + bias;
        }
    }
}

// ---------------- recurrence: 64 persistent WGs, per-step device barrier ----
__global__ __launch_bounds__(128) void k_recur(
    const float* __restrict__ gi, const bf16* __restrict__ whh,
    const float* __restrict__ bhh, const int* __restrict__ dones,
    float* __restrict__ hbuf, unsigned* __restrict__ bar,
    float* __restrict__ out, float* __restrict__ hlast) {
    __shared__ bf16  lds_h[32 * 512];   // 32 KB: A fragments
    __shared__ float gh[12 * 256];      // 12 KB: gh tiles
    __shared__ float gi_lds[16 * 192];  // 12 KB: async-staged gi tile [m][g*64+c]
    __shared__ float hcol[16 * 64];     //  4 KB: masked h, own column slice
    __shared__ float mask_lds[16];

    const int tid   = threadIdx.x;
    const int lane  = tid & 31;
    const int wv    = tid >> 5;
    const int bg    = blockIdx.x >> 4;
    const int cs    = blockIdx.x & 15;
    const int b0    = bg * 16;
    const int jbase = wv * 3;

    for (int t = 0; t < TT; ++t) {
        const float* hprev = hbuf + (size_t)(t & 1) * (BB * HH);
        float*       hnext = hbuf + (size_t)((t + 1) & 1) * (BB * HH);

        if (tid < 16)
            mask_lds[tid] = dones[(size_t)(b0 + tid) * TT + t] ? 0.0f : 1.0f;
        __syncthreads();

        // ---- stage masked h (16 x 1024): 8 elems/chunk, 16 chunks/thread ----
#pragma unroll 4
        for (int e = 0; e < 16; ++e) {
            int q = e * 128 + tid;           // 0..2047
            int m = q >> 7;                  // row; one row = 128 chunks
            int k = (q & 127) * 8;
            float mk = mask_lds[m];
            const float* hp = &hprev[(size_t)(b0 + m) * HH + k];
            v4f a0 = *(const v4f*)hp * mk;
            v4f a1 = *(const v4f*)(hp + 4) * mk;
            int kk = k & 31;
            int g  = kk >> 4;
            int j0 = kk & 8;
            v8bf pk;
#pragma unroll
            for (int u = 0; u < 4; ++u) { pk[u] = (bf16)a0[u]; pk[u + 4] = (bf16)a1[u]; }
            *(v8bf*)&lds_h[(k >> 5) * 512 + (m + 16 * g) * 16 + j0] = pk;
            if ((k >> 6) == cs) {           // keep fp32 copy of own slice
                *(v4f*)&hcol[m * 64 + (k & 63)]     = a0;
                *(v4f*)&hcol[m * 64 + (k & 63) + 4] = a1;
            }
        }
        __syncthreads();

        // ---- async-stage gi tile for this step (overlaps the WMMA loop) ----
        {
            const float* gbase = gi + (size_t)t * (BB * G3H) + cs * 64;
#pragma unroll
            for (int e = 0; e < 6; ++e) {
                int q   = e * 128 + tid;     // 0..767 b128 chunks
                int seg = q >> 4;            // 0..47 = m*3+g
                int c4  = q & 15;
                int m   = seg / 3;
                int g   = seg % 3;
                const float* gp = gbase + (size_t)(b0 + m) * G3H + g * HH + c4 * 4;
                unsigned lofs = (unsigned)(size_t)&gi_lds[m * 192 + g * 64 + c4 * 4];
                async_b128(lofs, gp);
            }
        }

        // ---- gh = h_masked @ Whh^T : 12 WMMA tiles (3/wave) ----
        v8f acc[3];
#pragma unroll
        for (int i = 0; i < 3; ++i) acc[i] = (v8f)0.0f;

        for (int k0 = 0; k0 < HH; k0 += 32) {
            v16bf afrag = *(const v16bf*)&lds_h[(k0 >> 5) * 512 + lane * 16];
#pragma unroll
            for (int i = 0; i < 3; ++i) {
                int j = jbase + i;
                int row0 = (j >> 2) * HH + cs * 64 + (j & 3) * 16;
                const bf16* bp = &whh[(size_t)(row0 + (lane & 15)) * HH
                                      + k0 + 16 * (lane >> 4)];
                if (k0 + 32 < HH) __builtin_prefetch(bp + 32, 0, 1);
                v16bf bfrag = *(const v16bf*)bp;
                acc[i] = __builtin_amdgcn_wmma_f32_16x16x32_bf16(
                    false, afrag, false, bfrag, (short)0, acc[i], false, false);
            }
        }

        wait_async0();   // gi tile now resident in LDS (own wave's part)

        // ---- spill gh (+ b_hh) to LDS ----
#pragma unroll
        for (int i = 0; i < 3; ++i) {
            int j = jbase + i;
            int row0 = (j >> 2) * HH + cs * 64 + (j & 3) * 16;
            float bias = bhh[row0 + (lane & 15)];
#pragma unroll
            for (int q = 0; q < 8; ++q)
                gh[j * 256 + (q + 8 * (lane >> 4)) * 16 + (lane & 15)] = acc[i][q] + bias;
        }
        __syncthreads();  // also publishes all waves' async gi chunks

        // ---- gate math: 16 rows x 64 cols, float4 per chunk ----
#pragma unroll
        for (int e = 0; e < 2; ++e) {
            int q  = e * 128 + tid;          // 0..255
            int m  = q >> 4;
            int c  = (q & 15) * 4;
            int jr = c >> 4;
            int cc = c & 15;
            v4f hr  = *(const v4f*)&gh[jr * 256 + m * 16 + cc];
            v4f hz  = *(const v4f*)&gh[(jr + 4) * 256 + m * 16 + cc];
            v4f hn  = *(const v4f*)&gh[(jr + 8) * 256 + m * 16 + cc];
            v4f ir  = *(const v4f*)&gi_lds[m * 192 + c];
            v4f iz  = *(const v4f*)&gi_lds[m * 192 + 64 + c];
            v4f in_ = *(const v4f*)&gi_lds[m * 192 + 128 + c];
            v4f hm  = *(const v4f*)&hcol[m * 64 + c];
            v4f hnew;
#pragma unroll
            for (int u = 0; u < 4; ++u) {
                float r = fast_sig(ir[u] + hr[u]);
                float z = fast_sig(iz[u] + hz[u]);
                float n = fast_tanh(in_[u] + r * hn[u]);
                hnew[u] = (1.0f - z) * n + z * hm[u];
            }
            int b = b0 + m;
            *(v4f*)&out[(size_t)b * (TT * HH) + (size_t)t * HH + cs * 64 + c] = hnew;
            *(v4f*)&hnext[(size_t)b * HH + cs * 64 + c] = hnew;
            if (t == TT - 1)
                *(v4f*)&hlast[(size_t)b * HH + cs * 64 + c] = hnew;
        }

        // ---- device-wide barrier (monotonic counter, reset by k_init) ----
        __threadfence();
        __syncthreads();
        if (tid == 0) {
            atomicAdd(bar, 1u);
            unsigned target = (unsigned)NWG_REC * (unsigned)(t + 1);
            while (__hip_atomic_load(bar, __ATOMIC_ACQUIRE,
                                     __HIP_MEMORY_SCOPE_AGENT) < target)
                __builtin_amdgcn_s_sleep(2);
        }
        __syncthreads();
    }
}

// ---------------- launch ----------------
extern "C" void kernel_launch(void* const* d_in, const int* in_sizes, int n_in,
                              void* d_out, int out_size, void* d_ws, size_t ws_size,
                              hipStream_t stream) {
    const float* x     = (const float*)d_in[0];
    const float* h0    = (const float*)d_in[1];
    const int*   dones = (const int*)d_in[2];
    const float* Wih   = (const float*)d_in[3];
    const float* Whh   = (const float*)d_in[4];
    const float* bih   = (const float*)d_in[5];
    const float* bhh   = (const float*)d_in[6];

    char* ws = (char*)d_ws;
    bf16*     wih_b = (bf16*)(ws + WS_WIH);
    bf16*     whh_b = (bf16*)(ws + WS_WHH);
    float*    hbuf  = (float*)(ws + WS_HBUF);
    unsigned* bar   = (unsigned*)(ws + WS_BAR);
    float*    gi    = (float*)(ws + WS_GI);

    float* out   = (float*)d_out;
    float* hlast = out + (size_t)BB * TT * HH;

    k_init<<<2048, 256, 0, stream>>>(Wih, Whh, h0, wih_b, whh_b, hbuf, bar);

    dim3 g1(G3H / 64, (BB * TT) / 64);   // (48, 512)
    k_gemm_gi<<<g1, 128, 0, stream>>>(x, wih_b, bih, gi);

    k_recur<<<NWG_REC, 128, 0, stream>>>(gi, whh_b, bhh, dones,
                                         hbuf, bar, out, hlast);
}
